// MultiHeadSelfAttention_29403346108551
// MI455X (gfx1250) — matrix operand order repaired and checked by reading, not compiled, not run on hardware
//
#include <hip/hip_runtime.h>
#include <hip/hip_bf16.h>

typedef __bf16 bf16_t;
typedef __attribute__((ext_vector_type(16))) __bf16 v16bf;
typedef __attribute__((ext_vector_type(8)))  __bf16 v8bf;
typedef __attribute__((ext_vector_type(8)))  float  v8f;

#define MODEL_DIM 1024
#define NUM_HEADS 16
#define D_K       64
#define BATCH     2
#define SEQ       2048
#define BT        (BATCH * SEQ)      // 4096 rows
#define ROT_ROW   SEQ                // reference indexes rot tables at position t == seq_len

// ---------------------------------------------------------------------------
// WMMA helpers (CDNA5: D = A(16x32 bf16) x B(32x16 bf16) + C(16x16 f32))
// ---------------------------------------------------------------------------
__device__ __forceinline__ v8f wmma_bf16(v16bf a, v16bf b, v8f c) {
  // 8 args: (neg_a, A, neg_b, B, c_mod, C, reuse_a, reuse_b)
  return __builtin_amdgcn_wmma_f32_16x16x32_bf16(false, a, false, b, (short)0, c,
                                                 false, false);
}

// A-matrix 16x32 bf16 layout: lane half h = lane>>4,
// per-lane elems 0..15 hold K = 16*h + 0..15 (contiguous, matches B).
// Caller passes p already offset by (row)*(stride) + k0 + 16*h.
__device__ __forceinline__ v16bf load_a16(const bf16_t* p) {
  v8bf lo = *(const v8bf*)(p);
  v8bf hi = *(const v8bf*)(p + 8);
  v16bf r;
#pragma unroll
  for (int i = 0; i < 8; ++i) { r[i] = lo[i]; r[i + 8] = hi[i]; }
  return r;
}

// ---------------------------------------------------------------------------
// fp32 -> bf16 conversion
// ---------------------------------------------------------------------------
__global__ __launch_bounds__(256) void f32_to_bf16_kernel(const float* __restrict__ src,
                                                          bf16_t* __restrict__ dst, int n) {
  int i = blockIdx.x * blockDim.x + threadIdx.x;
  if (i < n) dst[i] = (bf16_t)src[i];
}

// ---------------------------------------------------------------------------
// C[n,m] = sum_k A[n,k] * W[m,k] + bias[m]   (A,W bf16 row-major, C fp32)
// One wave computes a 64x32 tile (4x2 fragments, 8 WMMA per k-step).
// blockDim = 128 (4 waves).
// ---------------------------------------------------------------------------
__global__ __launch_bounds__(128) void gemm_nt_bias_kernel(
    const bf16_t* __restrict__ A, const bf16_t* __restrict__ W,
    const float* __restrict__ bias, float* __restrict__ C,
    int N, int K, int M) {
  const int lane = threadIdx.x & 31;
  const int wv   = threadIdx.x >> 5;
  const int tile = blockIdx.x * (blockDim.x >> 5) + wv;
  const int tilesM = M >> 5;                 // 32-wide column tiles
  const int tn = (tile / tilesM) << 6;       // 64-row tiles
  const int tm = (tile % tilesM) << 5;
  const int l16 = lane & 15;
  const int lh  = lane >> 4;

  v8f acc[4][2];
#pragma unroll
  for (int i = 0; i < 4; ++i)
#pragma unroll
    for (int j = 0; j < 2; ++j) acc[i][j] = (v8f){};

  const bf16_t* ap[4];
#pragma unroll
  for (int i = 0; i < 4; ++i)
    ap[i] = A + (size_t)(tn + 16 * i + l16) * K + lh * 16;
  const bf16_t* b0p = W + (size_t)(tm + l16) * K + lh * 16;
  const bf16_t* b1p = b0p + (size_t)16 * K;

  for (int k0 = 0; k0 < K; k0 += 32) {
    __builtin_prefetch(ap[0] + k0 + 256, 0, 3);   // near-cache prefetch, 512B ahead
    __builtin_prefetch(ap[2] + k0 + 256, 0, 3);
    __builtin_prefetch(b0p  + k0 + 256, 0, 3);
    v16bf b0 = *(const v16bf*)(b0p + k0);
    v16bf b1 = *(const v16bf*)(b1p + k0);
#pragma unroll
    for (int i = 0; i < 4; ++i) {
      v16bf a = load_a16(ap[i] + k0);
      acc[i][0] = wmma_bf16(a, b0, acc[i][0]);
      acc[i][1] = wmma_bf16(a, b1, acc[i][1]);
    }
  }

  const float bias0 = bias[tm + l16];
  const float bias1 = bias[tm + 16 + l16];
#pragma unroll
  for (int i = 0; i < 4; ++i) {
#pragma unroll
    for (int j = 0; j < 8; ++j) {
      const int r = tn + 16 * i + j + 8 * lh;   // D layout: M = j + 8*(l>>4)
      C[(size_t)r * M + tm + l16]      = acc[i][0][j] + bias0;
      C[(size_t)r * M + tm + 16 + l16] = acc[i][1][j] + bias1;
    }
  }
}

// ---------------------------------------------------------------------------
// Stage: bug-faithful head reshape + fixed-angle RoPE; emit bf16
//   q_r,k_r : [bh][t][64]   (roped)
//   v_t     : [bh][64][t]   (transposed for P*V B-fragments)
// Reference mapping: q[b,h,t,d] = Q[b, h*128 + t/16, (t%16)*64 + d]
// One thread handles one (bh, t, pair i) with i in 0..31 (d = 2i, 2i+1).
// ---------------------------------------------------------------------------
__global__ __launch_bounds__(256) void stage_rope_kernel(
    const float* __restrict__ Qf, const float* __restrict__ Kf,
    const float* __restrict__ Vf, const float* __restrict__ rot_cos,
    const float* __restrict__ rot_sin,
    bf16_t* __restrict__ q_r, bf16_t* __restrict__ k_r, bf16_t* __restrict__ v_t) {
  const int idx = blockIdx.x * blockDim.x + threadIdx.x;  // 32 bh * 2048 t * 32 i
  const int i  = idx & 31;
  const int t  = (idx >> 5) & (SEQ - 1);
  const int bh = idx >> 16;
  if (bh >= BATCH * NUM_HEADS) return;
  const int b = bh >> 4;
  const int h = bh & 15;

  const int src_row = b * SEQ + h * (SEQ * D_K / MODEL_DIM) + (t >> 4); // h*128 + t/16
  const size_t sbase = (size_t)src_row * MODEL_DIM + (size_t)(t & 15) * D_K;

  const float c = rot_cos[(size_t)ROT_ROW * (D_K / 2) + i];
  const float s = rot_sin[(size_t)ROT_ROW * (D_K / 2) + i];

  const float qe = Qf[sbase + 2 * i], qo = Qf[sbase + 2 * i + 1];
  const float ke = Kf[sbase + 2 * i], ko = Kf[sbase + 2 * i + 1];
  const float v0 = Vf[sbase + 2 * i], v1 = Vf[sbase + 2 * i + 1];

  const size_t dbase = ((size_t)bh * SEQ + t) * D_K;
  q_r[dbase + 2 * i]     = (bf16_t)(qe * c - qo * s);
  q_r[dbase + 2 * i + 1] = (bf16_t)(qe * s + qo * c);
  k_r[dbase + 2 * i]     = (bf16_t)(ke * c - ko * s);
  k_r[dbase + 2 * i + 1] = (bf16_t)(ke * s + ko * c);

  const size_t vbase = (size_t)bh * D_K * SEQ;
  v_t[vbase + (size_t)(2 * i)     * SEQ + t] = (bf16_t)v0;
  v_t[vbase + (size_t)(2 * i + 1) * SEQ + t] = (bf16_t)v1;
}

// ---------------------------------------------------------------------------
// Flash attention: one wave per (bh, 32-query tile) = two 16-row sub-tiles
// sharing every K/V fragment. Streams 32 keys/step: 8 WMMA for scores,
// online softmax, P through per-wave LDS to re-shape into A-fragment layout,
// 8 WMMA for P*V (V transposed). blockDim = 128 (4 waves).
// ---------------------------------------------------------------------------
__device__ __forceinline__ void softmax_update_store(
    v8f& s0, v8f& s1, float* mrow, float* lrow, v8f* oacc,
    bf16_t (*pl)[32], int lh, int l16) {
#pragma unroll
  for (int j = 0; j < 8; ++j) {
    float a   = s0[j] * 0.125f;     // 1/sqrt(64)
    float bvl = s1[j] * 0.125f;
    float mx = fmaxf(a, bvl);
#pragma unroll
    for (int msk = 1; msk < 16; msk <<= 1) mx = fmaxf(mx, __shfl_xor(mx, msk, 32));
    const float mnew  = fmaxf(mrow[j], mx);
    const float alpha = __expf(mrow[j] - mnew);
    mrow[j] = mnew;
    const float p0 = __expf(a - mnew);
    const float p1 = __expf(bvl - mnew);
    float ps = p0 + p1;
#pragma unroll
    for (int msk = 1; msk < 16; msk <<= 1) ps += __shfl_xor(ps, msk, 32);
    lrow[j] = lrow[j] * alpha + ps;
    oacc[0][j] *= alpha; oacc[1][j] *= alpha; oacc[2][j] *= alpha; oacc[3][j] *= alpha;
    const int r = j + 8 * lh;                 // D-layout row
    pl[r][l16]      = (bf16_t)p0;             // key cols 0..15
    pl[r][16 + l16] = (bf16_t)p1;             // key cols 16..31
  }
}

__global__ __launch_bounds__(128) void attention_kernel(
    const bf16_t* __restrict__ q_r, const bf16_t* __restrict__ k_r,
    const bf16_t* __restrict__ v_t, bf16_t* __restrict__ out_pre) {
  const int lane = threadIdx.x & 31;
  const int wv   = threadIdx.x >> 5;
  const int task = blockIdx.x * (blockDim.x >> 5) + wv;   // 0 .. 32*64-1
  const int bh  = task >> 6;                              // 32 (b,h) pairs
  const int tq0 = (task & 63) << 5;                       // 64 query tiles of 32
  const int l16 = lane & 15;
  const int lh  = lane >> 4;

  const bf16_t* q = q_r + (size_t)bh * SEQ * D_K;
  const bf16_t* k = k_r + (size_t)bh * SEQ * D_K;
  const bf16_t* v = v_t + (size_t)bh * D_K * SEQ;

  __shared__ __align__(16) bf16_t plds[4][2][16][32];     // 2 KB per wave
  bf16_t (*plA)[32] = plds[wv][0];
  bf16_t (*plB)[32] = plds[wv][1];

  // Q fragments: two 16-row sub-tiles x two d-halves, held for the whole sweep.
  const bf16_t* qpA = q + (size_t)(tq0 + l16) * D_K + lh * 16;
  const bf16_t* qpB = qpA + (size_t)16 * D_K;
  const v16bf qa0 = load_a16(qpA);
  const v16bf qa1 = load_a16(qpA + 32);
  const v16bf qb0 = load_a16(qpB);
  const v16bf qb1 = load_a16(qpB + 32);

  v8f oaccA[4], oaccB[4];
#pragma unroll
  for (int i = 0; i < 4; ++i) { oaccA[i] = (v8f){}; oaccB[i] = (v8f){}; }
  float mrowA[8], lrowA[8], mrowB[8], lrowB[8];
#pragma unroll
  for (int j = 0; j < 8; ++j) {
    mrowA[j] = -1e30f; lrowA[j] = 0.0f;
    mrowB[j] = -1e30f; lrowB[j] = 0.0f;
  }

  for (int tk0 = 0; tk0 < SEQ; tk0 += 32) {
    // ---- shared K fragments: two key sub-tiles x two d-halves ----
    const bf16_t* kp0 = k + (size_t)(tk0 + l16) * D_K + lh * 16;
    const bf16_t* kp1 = kp0 + (size_t)16 * D_K;
    const v16bf kb00 = *(const v16bf*)(kp0);        // keys 0-15,  d 0-31
    const v16bf kb01 = *(const v16bf*)(kp0 + 32);   // keys 0-15,  d 32-63
    const v16bf kb10 = *(const v16bf*)(kp1);        // keys 16-31, d 0-31
    const v16bf kb11 = *(const v16bf*)(kp1 + 32);   // keys 16-31, d 32-63

    v8f sA0 = {}, sA1 = {}, sB0 = {}, sB1 = {};
    sA0 = wmma_bf16(qa0, kb00, sA0);  sA0 = wmma_bf16(qa1, kb01, sA0);
    sA1 = wmma_bf16(qa0, kb10, sA1);  sA1 = wmma_bf16(qa1, kb11, sA1);
    sB0 = wmma_bf16(qb0, kb00, sB0);  sB0 = wmma_bf16(qb1, kb01, sB0);
    sB1 = wmma_bf16(qb0, kb10, sB1);  sB1 = wmma_bf16(qb1, kb11, sB1);

    softmax_update_store(sA0, sA1, mrowA, lrowA, oaccA, plA, lh, l16);
    softmax_update_store(sB0, sB1, mrowB, lrowB, oaccB, plB, lh, l16);

    // wave-local LDS fence: stores above must land before A-frag reload
    asm volatile("s_wait_dscnt 0" ::: "memory");

    const v16bf paA = load_a16(&plA[l16][lh * 16]);
    const v16bf paB = load_a16(&plB[l16][lh * 16]);

    // ---- shared V^T fragments feed both query sub-tiles ----
    const bf16_t* vp = v + (size_t)l16 * SEQ + tk0 + lh * 16;
#pragma unroll
    for (int nd = 0; nd < 4; ++nd) {
      const v16bf vb = *(const v16bf*)(vp + (size_t)(16 * nd) * SEQ);
      oaccA[nd] = wmma_bf16(paA, vb, oaccA[nd]);
      oaccB[nd] = wmma_bf16(paB, vb, oaccB[nd]);
    }
  }

  // ---- epilogue: normalize and scatter into [b, t, h*64+d] (bf16) ----
  const int b = bh >> 4;
  const int h = bh & 15;
#pragma unroll
  for (int j = 0; j < 8; ++j) {
    const float invA = 1.0f / lrowA[j];
    const float invB = 1.0f / lrowB[j];
    const int trA = tq0 + j + 8 * lh;
    const int trB = trA + 16;
    const size_t baseA = ((size_t)(b * SEQ + trA)) * MODEL_DIM + h * D_K;
    const size_t baseB = ((size_t)(b * SEQ + trB)) * MODEL_DIM + h * D_K;
#pragma unroll
    for (int nd = 0; nd < 4; ++nd) {
      out_pre[baseA + 16 * nd + l16] = (bf16_t)(oaccA[nd][j] * invA);
      out_pre[baseB + 16 * nd + l16] = (bf16_t)(oaccB[nd][j] * invB);
    }
  }
}

// ---------------------------------------------------------------------------
// Host-side orchestration
// ---------------------------------------------------------------------------
extern "C" void kernel_launch(void* const* d_in, const int* in_sizes, int n_in,
                              void* d_out, int out_size, void* d_ws, size_t ws_size,
                              hipStream_t stream) {
  const float* x  = (const float*)d_in[0];
  const float* wq = (const float*)d_in[1];
  const float* bq = (const float*)d_in[2];
  const float* wk = (const float*)d_in[3];
  const float* bk = (const float*)d_in[4];
  const float* wvp = (const float*)d_in[5];
  const float* bv = (const float*)d_in[6];
  const float* wo = (const float*)d_in[7];
  const float* bo = (const float*)d_in[8];
  const float* rc = (const float*)d_in[9];
  const float* rs = (const float*)d_in[10];

  size_t off = 0;
  auto take = [&](size_t bytes) -> void* {
    void* p = (char*)d_ws + off;
    off += (bytes + 255) & ~(size_t)255;
    return p;
  };

  const size_t NX = (size_t)BT * MODEL_DIM;          // 4096*1024
  const size_t NW = (size_t)MODEL_DIM * MODEL_DIM;   // 1024*1024
  const size_t NH = (size_t)BATCH * NUM_HEADS * SEQ * D_K;

  bf16_t* xb  = (bf16_t*)take(NX * sizeof(bf16_t));
  bf16_t* wqb = (bf16_t*)take(NW * sizeof(bf16_t));
  bf16_t* wkb = (bf16_t*)take(NW * sizeof(bf16_t));
  bf16_t* wvb = (bf16_t*)take(NW * sizeof(bf16_t));
  bf16_t* wob = (bf16_t*)take(NW * sizeof(bf16_t));
  float*  Qf  = (float*)take(NX * sizeof(float));
  float*  Kf  = (float*)take(NX * sizeof(float));
  float*  Vf  = (float*)take(NX * sizeof(float));
  bf16_t* qr  = (bf16_t*)take(NH * sizeof(bf16_t));
  bf16_t* kr  = (bf16_t*)take(NH * sizeof(bf16_t));
  bf16_t* vt  = (bf16_t*)take(NH * sizeof(bf16_t));
  bf16_t* opre = (bf16_t*)take(NX * sizeof(bf16_t));

  // 1) fp32 -> bf16
  f32_to_bf16_kernel<<<(int)((NX + 255) / 256), 256, 0, stream>>>(x, xb, (int)NX);
  f32_to_bf16_kernel<<<(int)((NW + 255) / 256), 256, 0, stream>>>(wq, wqb, (int)NW);
  f32_to_bf16_kernel<<<(int)((NW + 255) / 256), 256, 0, stream>>>(wk, wkb, (int)NW);
  f32_to_bf16_kernel<<<(int)((NW + 255) / 256), 256, 0, stream>>>(wvp, wvb, (int)NW);
  f32_to_bf16_kernel<<<(int)((NW + 255) / 256), 256, 0, stream>>>(wo, wob, (int)NW);

  // 2) QKV projections (4096x1024 @ 1024x1024^T), 64x32 tile per wave
  const int gemm_tiles  = (BT / 64) * (MODEL_DIM / 32);  // 2048
  const int gemm_blocks = gemm_tiles / 4;                // 4 waves per block
  gemm_nt_bias_kernel<<<gemm_blocks, 128, 0, stream>>>(xb, wqb, bq, Qf, BT, MODEL_DIM, MODEL_DIM);
  gemm_nt_bias_kernel<<<gemm_blocks, 128, 0, stream>>>(xb, wkb, bk, Kf, BT, MODEL_DIM, MODEL_DIM);
  gemm_nt_bias_kernel<<<gemm_blocks, 128, 0, stream>>>(xb, wvb, bv, Vf, BT, MODEL_DIM, MODEL_DIM);

  // 3) reshape + RoPE + V-transpose staging
  const size_t stage_threads = (size_t)BATCH * NUM_HEADS * SEQ * (D_K / 2); // 2M
  stage_rope_kernel<<<(int)(stage_threads / 256), 256, 0, stream>>>(Qf, Kf, Vf, rc, rs, qr, kr, vt);

  // 4) attention: 32 bh * 64 query-tiles(32 rows) = 2048 wave tasks, 4 waves/block
  attention_kernel<<<512, 128, 0, stream>>>(qr, kr, vt, opre);

  // 5) output projection -> fp32 d_out
  gemm_nt_bias_kernel<<<gemm_blocks, 128, 0, stream>>>(opre, wob, bo, (float*)d_out,
                                                       BT, MODEL_DIM, MODEL_DIM);
}
